// Model_10728828305676
// MI455X (gfx1250) — matrix operand order repaired and checked by reading, not compiled, not run on hardware
//
#include <hip/hip_runtime.h>
#include <hip/hip_bf16.h>

// RNN: B=2048, T=64, I=8, H=32, L=100.
// One wave32 per workgroup owns a 16-row batch tile; the full T x 16 x 32
// activation sequence lives in LDS (f16, 64KB) and is updated in place layer
// by layer. Per timestep: D(16x32) = tanh([x_t | h] @ [Wih^T ; Whh^T] + bias)
// via 4x v_wmma_f32_16x16x32_f16 (K=32 chunks, 2 column tiles).
// tanh is branchless (native v_tanh_f32 if available, else exp2+rcp).

typedef __attribute__((ext_vector_type(16))) _Float16 v16h;
typedef __attribute__((ext_vector_type(8)))  _Float16 v8h;
typedef __attribute__((ext_vector_type(8)))  float    v8f;

#define RNN_B  2048
#define RNN_T  64
#define RNN_I  8
#define RNN_H  32
#define RNN_L  100
#define RNN_MT 16   // batch rows per wave / workgroup

__device__ __forceinline__ float fast_tanh(float x) {
#if __has_builtin(__builtin_amdgcn_tanhf)
    return __builtin_amdgcn_tanhf(x);          // gfx1250 v_tanh_f32
#else
    // branchless: tanh(x) = sign(x) * (1 - 2/(1 + exp2(2*log2(e)*|x|)))
    // large |x|: exp2 -> +inf, rcp -> 0, result -> +-1 (correct saturation)
    const float ax = __builtin_fabsf(x);
    const float e  = __builtin_amdgcn_exp2f(ax * 2.8853900817779268f);
    const float r  = 1.0f - 2.0f * __builtin_amdgcn_rcpf(1.0f + e);
    return __builtin_copysignf(r, x);
#endif
}

__global__ __launch_bounds__(32)
void rnn100_wmma_kernel(const float* __restrict__ x,
                        const float* __restrict__ Wih0,
                        const float* __restrict__ Whh0,
                        const float* __restrict__ bih0,
                        const float* __restrict__ bhh0,
                        const float* __restrict__ Wih,
                        const float* __restrict__ Whh,
                        const float* __restrict__ bih,
                        const float* __restrict__ bhh,
                        const float* __restrict__ Wout,
                        const float* __restrict__ bout,
                        float* __restrict__ out)
{
    // T x MT x H halfs = 64KB. Row-major [t][m][k].
    __shared__ _Float16 seq[RNN_T * RNN_MT * RNN_H];

    const int lane   = threadIdx.x;       // 0..31 (wave32)
    const int nlo    = lane & 15;         // column / row-in-tile selector
    const int hiHalf = lane >> 4;         // 0: lanes 0-15, 1: lanes 16-31
    const int b0     = blockIdx.x * RNN_MT;

    // ---- stage layer-0 input into LDS: cols 0..7 = x, cols 8..31 = 0 ------
    for (int p = lane; p < RNN_T * RNN_MT; p += 32) {
        const int t = p / RNN_MT, m = p % RNN_MT;
        const float* xp = x + ((size_t)(b0 + m) * RNN_T + t) * RNN_I;
        _Float16* s = &seq[(t * RNN_MT + m) * RNN_H];
        #pragma unroll
        for (int k = 0; k < RNN_I; ++k) s[k] = (_Float16)xp[k];
        #pragma unroll
        for (int k = RNN_I; k < RNN_H; ++k) s[k] = (_Float16)0.0f;
    }
    __syncthreads();

    for (int l = 0; l < RNN_L; ++l) {
        // hint next layer's weights toward the caches (global_prefetch_b8)
        if (l + 1 < RNN_L) {
            __builtin_prefetch(Wih + (size_t)l * RNN_H * RNN_H, 0, 1);
            __builtin_prefetch(Whh + (size_t)l * RNN_H * RNN_H, 0, 1);
        }

        const float* Wi = (l == 0) ? Wih0 : (Wih + (size_t)(l - 1) * RNN_H * RNN_H);
        const float* Wh = (l == 0) ? Whh0 : (Whh + (size_t)(l - 1) * RNN_H * RNN_H);
        const float* bi = (l == 0) ? bih0 : (bih + (size_t)(l - 1) * RNN_H);
        const float* bh = (l == 0) ? bhh0 : (bhh + (size_t)(l - 1) * RNN_H);
        const int in_dim = (l == 0) ? RNN_I : RNN_H;

        // ---- B-operands (32x16 f16 each): lane n = lane%16 (+16*j tile),
        //      half p <-> K = hiHalf*8 + p (p<8), hiHalf*8 + 8 + p (p>=8). Wcomb[k][n] = W[n][k].
        //      Bias folded into accumulator init (same value in all 8 slots).
        const int kb = hiHalf * 16;
        v16h Bx[2], Bh[2];
        v8f  Cb[2];
        #pragma unroll
        for (int j = 0; j < 2; ++j) {
            const int n = nlo + 16 * j;
            const float bv = bi[n] + bh[n];
            #pragma unroll
            for (int e = 0; e < 8; ++e) Cb[j][e] = bv;
            #pragma unroll
            for (int p = 0; p < 16; ++p) {
                const int k = hiHalf * 8 + (p < 8 ? p : p + 8);
                Bx[j][p] = (k < in_dim) ? (_Float16)Wi[n * in_dim + k]
                                        : (_Float16)0.0f;
                Bh[j][p] = (_Float16)Wh[n * RNN_H + k];
            }
        }

        // h(t=-1) = 0
        v16h Ah;
        #pragma unroll
        for (int e = 0; e < 16; ++e) Ah[e] = (_Float16)0.0f;

        for (int t = 0; t < RNN_T; ++t) {
            // ---- A-operand (16x32 f16) for x_t from LDS row m = lane%16:
            //      halves 0..7  <-> K = hiHalf*8 .. +7
            //      halves 8..15 <-> K = hiHalf*8+16 .. +23
            const _Float16* srow = &seq[(t * RNN_MT + nlo) * RNN_H];
            v8h a_lo = *(const v8h*)(srow + hiHalf * 8);
            v8h a_hi = *(const v8h*)(srow + hiHalf * 8 + 16);
            v16h Ax;
            #pragma unroll
            for (int e = 0; e < 8; ++e) { Ax[e] = a_lo[e]; Ax[8 + e] = a_hi[e]; }

            v8f acc0 = Cb[0], acc1 = Cb[1];
            acc0 = __builtin_amdgcn_wmma_f32_16x16x32_f16(false, Ax, false, Bx[0],
                                                          (short)0, acc0, false, false);
            acc0 = __builtin_amdgcn_wmma_f32_16x16x32_f16(false, Ah, false, Bh[0],
                                                          (short)0, acc0, false, false);
            acc1 = __builtin_amdgcn_wmma_f32_16x16x32_f16(false, Ax, false, Bx[1],
                                                          (short)0, acc1, false, false);
            acc1 = __builtin_amdgcn_wmma_f32_16x16x32_f16(false, Ah, false, Bh[1],
                                                          (short)0, acc1, false, false);

            // ---- h = tanh(acc); write back row-major into seq[t].
            //      D layout: vgpr v -> row m = v + hiHalf*8, col n = nlo (+16).
            _Float16* dst = &seq[t * RNN_MT * RNN_H];
            #pragma unroll
            for (int v = 0; v < 8; ++v) {
                const int m = v + hiHalf * 8;
                const float h0 = fast_tanh(acc0[v]);
                const float h1 = fast_tanh(acc1[v]);
                dst[m * RNN_H + nlo]      = (_Float16)h0;
                dst[m * RNN_H + nlo + 16] = (_Float16)h1;
                acc0[v] = h0;
                acc1[v] = h1;
            }
            __syncthreads();  // single-wave WG: LDS ops are in-order; cheap

            // reload h(t) as next step's A-operand (lane-transpose via LDS)
            v8h h_lo = *(const v8h*)(srow + hiHalf * 8);
            v8h h_hi = *(const v8h*)(srow + hiHalf * 8 + 16);
            #pragma unroll
            for (int e = 0; e < 8; ++e) { Ah[e] = h_lo[e]; Ah[8 + e] = h_hi[e]; }

            // ---- final projection: out[b] = h_last . Wout + bout ---------
            if (l == RNN_L - 1 && t == RNN_T - 1) {
                // alias dead t=0 region of seq as f32 reduction scratch
                float* red = (float*)seq;
                const float w0 = Wout[nlo];
                const float w1 = Wout[nlo + 16];
                if (lane < RNN_MT) red[lane] = 0.0f;
                __syncthreads();
                #pragma unroll
                for (int v = 0; v < 8; ++v) {
                    const int m = v + hiHalf * 8;
                    atomicAdd(&red[m], acc0[v] * w0 + acc1[v] * w1);  // ds_add_f32
                }
                __syncthreads();
                if (lane < RNN_MT) out[b0 + lane] = red[lane] + bout[0];
            }
        }
        __syncthreads();
    }
}

extern "C" void kernel_launch(void* const* d_in, const int* in_sizes, int n_in,
                              void* d_out, int out_size, void* d_ws, size_t ws_size,
                              hipStream_t stream) {
    (void)in_sizes; (void)n_in; (void)d_ws; (void)ws_size; (void)out_size;
    const float* x    = (const float*)d_in[0];
    const float* Wih0 = (const float*)d_in[1];
    const float* Whh0 = (const float*)d_in[2];
    const float* bih0 = (const float*)d_in[3];
    const float* bhh0 = (const float*)d_in[4];
    const float* Wih  = (const float*)d_in[5];
    const float* Whh  = (const float*)d_in[6];
    const float* bih  = (const float*)d_in[7];
    const float* bhh  = (const float*)d_in[8];
    const float* Wout = (const float*)d_in[9];
    const float* bout = (const float*)d_in[10];
    float* out = (float*)d_out;

    dim3 grid(RNN_B / RNN_MT);   // 128 independent batch tiles
    dim3 block(32);              // one wave32 per tile
    rnn100_wmma_kernel<<<grid, block, 0, stream>>>(
        x, Wih0, Whh0, bih0, bhh0, Wih, Whh, bih, bhh, Wout, bout, out);
}
